// MultiTokenAttention_8521215115368
// MI455X (gfx1250) — compile-verified, matrix operand order checked by reading, not run on hardware
//
#include <hip/hip_runtime.h>
#include <hip/hip_bf16.h>
#include <math.h>

#define S_LEN 2048
#define HEADS 16
#define DHEAD 64
#define GRP   8
#define HKSZ  2
#define BM    32      // query rows per block
#define BN    64      // key cols per tile
#define SROWS 48      // BM + query-conv halo (2+3), padded to 16
#define SCOLS 80      // BN + key-conv halo (5+5), padded to 16
#define QKS   6
#define KKS   11
#define EPSV  1e-5f

typedef __attribute__((ext_vector_type(16))) __bf16 v16bf;
typedef __attribute__((ext_vector_type(8)))  float  v8f;

union Frag { v16bf v; unsigned u[8]; };

__device__ inline unsigned short f2bf(float f) {
  unsigned u = __builtin_bit_cast(unsigned, f);
  unsigned r = (u + 0x7FFFu + ((u >> 16) & 1u)) >> 16;   // round-nearest-even
  return (unsigned short)r;
}
__device__ inline unsigned pack2(float lo, float hi) {
  return (unsigned)f2bf(lo) | ((unsigned)f2bf(hi) << 16);
}

// A/B fragment in CDNA5 16-bit WMMA layout, sourced from global f32 (+zero mask).
// p = row base + half*8 already applied; pairs at {0,2,4,6} and {16,18,20,22}.
__device__ inline v16bf frag_from_global(const float* __restrict__ p, float msk) {
  Frag f;
#pragma unroll
  for (int i = 0; i < 4; ++i) {
    f.u[i]     = pack2(p[2*i] * msk,      p[2*i+1] * msk);
    f.u[4 + i] = pack2(p[16 + 2*i] * msk, p[16 + 2*i + 1] * msk);
  }
  return f.v;
}

__device__ inline v16bf frag_from_lds(const unsigned short* p) {
  Frag f;
#pragma unroll
  for (int i = 0; i < 4; ++i) {
    f.u[i]     = *(const unsigned*)(p + 2*i);
    f.u[4 + i] = *(const unsigned*)(p + 16 + 2*i);
  }
  return f.v;
}

__device__ inline v8f wmma_bf16(v16bf a, v16bf b, v8f c) {
  return __builtin_amdgcn_wmma_f32_16x16x32_bf16(false, a, false, b, (short)0, c,
                                                 false, false);
}

__global__ __launch_bounds__(256)
void mta_fused_kernel(const float* __restrict__ q, const float* __restrict__ k,
                      const float* __restrict__ v, const float* __restrict__ cw,
                      const float* __restrict__ cb, const float* __restrict__ hm,
                      const float* __restrict__ gam, const float* __restrict__ bet,
                      float* __restrict__ out)
{
  const int q0   = blockIdx.x * BM;
  const int g    = blockIdx.y;
  const int tid  = threadIdx.x;
  const int lane = tid & 31;
  const int wvid = tid >> 5;
  const int m    = lane & 15;
  const int half = lane >> 4;

  // LDS: score panel (f32) unioned with final per-group output buffer
  __shared__ union { float sm[2][SROWS][SCOLS]; float o[2][BM][DHEAD]; } uSM; // 30.7 KB
  __shared__ unsigned short vt[2][DHEAD][BN];    // V transposed, bf16: 16 KB
  __shared__ unsigned short pbuf[2][BM][BN];     // P tile, bf16: 8 KB
  __shared__ float mrow[2][BM], lrow[2][BM], rsc[2][BM];
  __shared__ float wconv[2][QKS * KKS];
  __shared__ float bconv[2];

  // ---- init running softmax stats + conv weights ----
  for (int i = tid; i < 2 * BM; i += 256) {
    (&mrow[0][0])[i] = -1e30f;
    (&lrow[0][0])[i] = 0.f;
  }
  for (int i = tid; i < 2 * QKS * KKS; i += 256)
    (&wconv[0][0])[i] = cw[(size_t)(g * HKSZ + i / (QKS * KKS)) * QKS * KKS + i % (QKS * KKS)];
  if (tid < 2) bconv[tid] = cb[g * HKSZ + tid];

  // PV wave role: combo (score-head hi, V-head hj) x row-block rb2
  const int c4 = wvid & 3, hi = c4 & 1, hj = c4 >> 1, rb2 = wvid >> 2;
  v8f acc[4];
#pragma unroll
  for (int nb = 0; nb < 4; ++nb) acc[nb] = (v8f){0.f,0.f,0.f,0.f,0.f,0.f,0.f,0.f};

  const float scale = 0.125f;                 // 1/sqrt(64)
  const int ktmax = (q0 + BM - 1) / BN;       // causal: keys only up to q0+BM-1
  __syncthreads();

  for (int kt = 0; kt <= ktmax; ++kt) {
    const int k0 = kt * BN;

    // ---- (a) stage V tile transposed into LDS as bf16 (B-layout needs key-contig) ----
    for (int idx = tid; idx < 2 * BN * (DHEAD / 4); idx += 256) {
      int head = idx / (BN * (DHEAD / 4));
      int rem  = idx % (BN * (DHEAD / 4));
      int row  = rem / (DHEAD / 4);            // key within tile (always < S here)
      int d4   = (rem % (DHEAD / 4)) * 4;
      int gh   = g * HKSZ + head;
      const float4 vv = *(const float4*)(v + ((size_t)gh * S_LEN + (k0 + row)) * DHEAD + d4);
      vt[head][d4 + 0][row] = f2bf(vv.x);
      vt[head][d4 + 1][row] = f2bf(vv.y);
      vt[head][d4 + 2][row] = f2bf(vv.z);
      vt[head][d4 + 3][row] = f2bf(vv.w);
    }
    __syncthreads();

    // ---- (b) score panel: 2 heads x 3x5 WMMA tiles, masked+scaled to LDS f32 ----
    for (int t = wvid; t < 30; t += 8) {
      int head = t / 15, tt = t % 15, rb = tt / 5, nb = tt % 5;
      int gh = g * HKSZ + head;
      int qrow = q0 - 2 + rb * 16 + m;
      int qrc  = min(max(qrow, 0), S_LEN - 1);
      float qm = (qrow >= 0 && qrow < S_LEN) ? 1.f : 0.f;
      int keyr = k0 - 5 + nb * 16 + m;
      int krc  = min(max(keyr, 0), S_LEN - 1);
      float km = (keyr >= 0 && keyr < S_LEN) ? 1.f : 0.f;
      const float* qp = q + ((size_t)gh * S_LEN + qrc) * DHEAD + half * 8;
      const float* kp = k + ((size_t)gh * S_LEN + krc) * DHEAD + half * 8;
      v8f c = (v8f){0.f,0.f,0.f,0.f,0.f,0.f,0.f,0.f};
      c = wmma_bf16(frag_from_global(qp, qm),      frag_from_global(kp, km),      c);
      c = wmma_bf16(frag_from_global(qp + 32, qm), frag_from_global(kp + 32, km), c);
#pragma unroll
      for (int r = 0; r < 8; ++r) {                 // C layout: M=r+8*half, N=m
        int M  = r + half * 8;
        int qg = q0 - 2 + rb * 16 + M;
        int kg = k0 - 5 + nb * 16 + m;
        bool ok = (qg >= 0) && (qg < S_LEN) && (kg >= 0) && (kg <= qg);
        uSM.sm[head][rb * 16 + M][nb * 16 + m] = ok ? c[r] * scale : 0.f;
      }
    }
    __syncthreads();

    // ---- (c) 6x11 depthwise conv + bias + causal(-1e9) + online softmax ----
    {
      int rowid = tid >> 2;          // 64 rows = 2 heads x 32 queries
      int head  = rowid >> 5;
      int qq    = rowid & 31;
      int sub   = tid & 3;           // 4 lanes per row, 16 key-cols each
      int c0    = sub * 16;
      float accv[16];
      float bias = bconv[head];
#pragma unroll
      for (int j = 0; j < 16; ++j) accv[j] = bias;
#pragma unroll
      for (int dq = 0; dq < QKS; ++dq) {
        const float* rp = &uSM.sm[head][qq + dq][c0];
        float seg[26];
#pragma unroll
        for (int x = 0; x < 26; ++x) seg[x] = rp[x];
#pragma unroll
        for (int dk = 0; dk < KKS; ++dk) {
          float wv2 = wconv[head][dq * KKS + dk];
#pragma unroll
          for (int j = 0; j < 16; ++j)
            accv[j] = fmaf(wv2, seg[j + dk], accv[j]);
        }
      }
      int qg = q0 + qq;
      float tmax = -1e30f;
#pragma unroll
      for (int j = 0; j < 16; ++j) {
        if (k0 + c0 + j > qg) accv[j] = -1e9f;
        tmax = fmaxf(tmax, accv[j]);
      }
      tmax = fmaxf(tmax, __shfl_xor(tmax, 1, 32));
      tmax = fmaxf(tmax, __shfl_xor(tmax, 2, 32));
      float mold = mrow[head][qq];
      float mnew = fmaxf(mold, tmax);
      float sfac = __expf(mold - mnew);
      float psum = 0.f;
#pragma unroll
      for (int j = 0; j < 16; ++j) {
        float p = __expf(accv[j] - mnew);
        psum += p;
        pbuf[head][qq][c0 + j] = f2bf(p);
      }
      psum += __shfl_xor(psum, 1, 32);
      psum += __shfl_xor(psum, 2, 32);
      if (sub == 0) {
        mrow[head][qq] = mnew;
        lrow[head][qq] = lrow[head][qq] * sfac + psum;
        rsc[head][qq]  = sfac;
      }
    }
    __syncthreads();

    // ---- (d) rescale accumulators + P·V WMMA (A from pbuf, B from vt) ----
    {
#pragma unroll
      for (int r = 0; r < 8; ++r) {
        float f = rsc[hi][rb2 * 16 + r + half * 8];
#pragma unroll
        for (int nb = 0; nb < 4; ++nb) acc[nb][r] *= f;
      }
      const unsigned short* pa = &pbuf[hi][rb2 * 16 + m][half * 8];
      v16bf a0 = frag_from_lds(pa);
      v16bf a1 = frag_from_lds(pa + 32);
#pragma unroll
      for (int nb = 0; nb < 4; ++nb) {
        const unsigned short* pb = &vt[hj][nb * 16 + m][half * 8];
        acc[nb] = wmma_bf16(a0, frag_from_lds(pb),      acc[nb]);
        acc[nb] = wmma_bf16(a1, frag_from_lds(pb + 32), acc[nb]);
      }
    }
    __syncthreads();
  } // key tiles

  // ---- head mixing: out_j = sum_i mix[g,i,j] * A_ij / l_i ----
  const float mixv = hm[(size_t)(g * HKSZ + hi) * HKSZ + hj];
  if (hi == 0) {
#pragma unroll
    for (int r = 0; r < 8; ++r) {
      int row = rb2 * 16 + r + half * 8;
      float inv = mixv / lrow[0][row];
#pragma unroll
      for (int nb = 0; nb < 4; ++nb)
        uSM.o[hj][row][nb * 16 + m] = acc[nb][r] * inv;
    }
  }
  __syncthreads();
  if (hi == 1) {
#pragma unroll
    for (int r = 0; r < 8; ++r) {
      int row = rb2 * 16 + r + half * 8;
      float inv = mixv / lrow[1][row];
#pragma unroll
      for (int nb = 0; nb < 4; ++nb)
        uSM.o[hj][row][nb * 16 + m] += acc[nb][r] * inv;
    }
  }
  __syncthreads();

  // ---- LayerNorm over D=64 + store ----
  {
    int rowid = tid >> 2;
    int j   = rowid >> 5;
    int qq  = rowid & 31;
    int sub = tid & 3;
    int c0  = sub * 16;
    float xs[16];
    float s1 = 0.f, s2 = 0.f;
#pragma unroll
    for (int t2 = 0; t2 < 16; ++t2) {
      float x = uSM.o[j][qq][c0 + t2];
      xs[t2] = x; s1 += x; s2 += x * x;
    }
    s1 += __shfl_xor(s1, 1, 32); s2 += __shfl_xor(s2, 1, 32);
    s1 += __shfl_xor(s1, 2, 32); s2 += __shfl_xor(s2, 2, 32);
    float mean = s1 * (1.f / 64.f);
    float var  = s2 * (1.f / 64.f) - mean * mean;
    float rstd = rsqrtf(var + EPSV);
    int gh = g * HKSZ + j;
    float* op = out + (((size_t)gh * S_LEN) + (q0 + qq)) * DHEAD + c0;
#pragma unroll
    for (int t2 = 0; t2 < 16; ++t2) {
      int d = c0 + t2;
      op[t2] = (xs[t2] - mean) * rstd * gam[d] + bet[d];
    }
  }
}

extern "C" void kernel_launch(void* const* d_in, const int* in_sizes, int n_in,
                              void* d_out, int out_size, void* d_ws, size_t ws_size,
                              hipStream_t stream) {
  const float* q  = (const float*)d_in[0];
  const float* k  = (const float*)d_in[1];
  const float* v  = (const float*)d_in[2];
  const float* cw = (const float*)d_in[3];
  const float* cb = (const float*)d_in[4];
  const float* hm = (const float*)d_in[5];
  const float* gm = (const float*)d_in[6];
  const float* bt = (const float*)d_in[7];
  (void)in_sizes; (void)n_in; (void)d_ws; (void)ws_size; (void)out_size;
  dim3 grid(S_LEN / BM, GRP, 1);
  mta_fused_kernel<<<grid, 256, 0, stream>>>(q, k, v, cw, cb, hm, gm, bt,
                                             (float*)d_out);
}
